// DateParser_1099511628256
// MI455X (gfx1250) — matrix operand order repaired and checked by reading, not compiled, not run on hardware
//
#include <hip/hip_runtime.h>

#define TXn  30
#define TYn  10
#define VIN  37
#define VOUT 11
#define Dn   32
#define Hn   64
#define An   10
#define Bn   16384

typedef _Float16 h16 __attribute__((ext_vector_type(16)));
typedef _Float16 h8  __attribute__((ext_vector_type(8)));
typedef float    f8  __attribute__((ext_vector_type(8)));

union H16u { h16 v; h8 h[2]; };

// A-fragment (16x32 f16, M x K) from LDS row-major [16][strideH] f16.
// ISA layout: lanes 0-15 = rows M, elems 0..7 -> K=kb..kb+7, elems 8..15 -> K=kb+16..kb+23;
// lanes 16-31 get +8 on K.  Two 16B loads per lane.
__device__ __forceinline__ h16 frag_a_lds(const _Float16* buf, int strideH, int kbase, int lane) {
  int m  = lane & 15;
  int th = lane >> 4;
  const _Float16* p = buf + m * strideH + kbase + th * 8;
  H16u r;
  r.h[0] = *(const h8*)(p);
  r.h[1] = *(const h8*)(p + 16);
  return r.v;
}

// Same A-fragment gather from GLOBAL memory; caller supplies this lane's row
// pointer (row m = lane&15 already folded in).  Rows stay hot in WGP$.
__device__ __forceinline__ h16 frag_a_glb(const _Float16* rowp, int kbase, int lane) {
  int th = lane >> 4;
  const _Float16* p = rowp + kbase + th * 8;
  H16u r;
  r.h[0] = *(const h8*)(p);
  r.h[1] = *(const h8*)(p + 16);
  return r.v;
}

// B-fragment (32x16 f16, K x N): column n = row of weight matrix W[n][k].
// Same K order as A: elems 0..7 -> K=kb+8*th.., elems 8..15 -> K=kb+16+8*th..
// Two 16B global loads per lane; weights stay hot in WGP$.
__device__ __forceinline__ h16 frag_b_glb(const _Float16* wrow, int kbase, int lane) {
  int th = lane >> 4;
  const _Float16* p = wrow + kbase + th * 8;
  H16u r;
  r.h[0] = *(const h8*)(p);
  r.h[1] = *(const h8*)(p + 16);
  return r.v;
}

__device__ __forceinline__ f8 wmma16(h16 a, h16 b, f8 c) {
  return __builtin_amdgcn_wmma_f32_16x16x32_f16(false, a, false, b, (short)0, c, false, false);
}

// Fast activations: native v_tanh_f32 (confirmed on gfx1250) + v_exp/v_rcp.
__device__ __forceinline__ float sigf(float x) {
  return __builtin_amdgcn_rcpf(1.0f + __expf(-x));
}
__device__ __forceinline__ float tanh_fast(float x) {
#if __has_builtin(__builtin_amdgcn_tanhf)
  return __builtin_amdgcn_tanhf(x);
#else
  float e = __expf(2.0f * x);
  return 1.0f - 2.0f * __builtin_amdgcn_rcpf(e + 1.0f);
#endif
}

// ---------------------------------------------------------------------------
// K0a: pack weights to padded row-major f16 (K padded to multiples of 32)
// ---------------------------------------------------------------------------
__global__ void __launch_bounds__(256) pack_weights_kernel(
    const float* __restrict__ wihf, const float* __restrict__ whhf,
    const float* __restrict__ wihr, const float* __restrict__ whhr,
    const float* __restrict__ wihp, const float* __restrict__ whhp,
    const float* __restrict__ wa1,
    _Float16* __restrict__ Wf_ih, _Float16* __restrict__ Wf_hh,
    _Float16* __restrict__ Wr_ih, _Float16* __restrict__ Wr_hh,
    _Float16* __restrict__ Wp,
    _Float16* __restrict__ Wa1x, _Float16* __restrict__ Wa1s)
{
  int t = threadIdx.x;
  for (int i = t; i < 128 * 64; i += 256) {       // [4D][64], src [4D][37]
    int n = i >> 6, k = i & 63;
    Wf_ih[i] = (_Float16)(k < VIN ? wihf[n * VIN + k] : 0.0f);
    Wr_ih[i] = (_Float16)(k < VIN ? wihr[n * VIN + k] : 0.0f);
  }
  for (int i = t; i < 128 * 32; i += 256) {       // [4D][32]
    Wf_hh[i] = (_Float16)whhf[i];
    Wr_hh[i] = (_Float16)whhr[i];
  }
  for (int i = t; i < 256 * 128; i += 256) {      // [4H][128] = [w_ih_p | w_hh_p]
    int n = i >> 7, k = i & 127;
    Wp[i] = (_Float16)(k < 64 ? wihp[n * 64 + k] : whhp[n * 64 + (k - 64)]);
  }
  for (int i = t; i < 16 * 64; i += 256) {        // attention mlp halves, rows padded to 16
    int a = i >> 6, k = i & 63;
    Wa1x[i] = (_Float16)((a < An) ? wa1[a * (2 * Hn) + k] : 0.0f);
    Wa1s[i] = (_Float16)((a < An) ? wa1[a * (2 * Hn) + 64 + k] : 0.0f);
  }
}

// ---------------------------------------------------------------------------
// K0b: X (f32, [B][TX][37]) -> X16 (f16, [B][TX][64], zero-padded K)
// so the encoder can gather x A-fragments straight from global memory.
// ---------------------------------------------------------------------------
__global__ void __launch_bounds__(256) convert_x_kernel(
    const float* __restrict__ X, _Float16* __restrict__ X16)
{
  size_t i = (size_t)blockIdx.x * 256 + threadIdx.x;   // index into [B*TX][64]
  int    k   = (int)(i & 63);
  size_t row = i >> 6;
  X16[i] = (_Float16)((k < VIN) ? X[row * VIN + k] : 0.0f);
}

// ---------------------------------------------------------------------------
// K1: bidirectional pre-LSTM encoder.  grid=(128,2) blocks of 256 (8 waves).
// Each wave owns 16 batch rows for all 30 time steps; c-state in registers,
// x A-fragments gathered directly from X16 (no staging), h via 8KB LDS.
// ---------------------------------------------------------------------------
__global__ void __launch_bounds__(256) prelstm_kernel(
    const _Float16* __restrict__ X16,
    const float* __restrict__ b_ih_f, const float* __restrict__ b_hh_f,
    const float* __restrict__ b_ih_r, const float* __restrict__ b_hh_r,
    const _Float16* __restrict__ Wf_ih, const _Float16* __restrict__ Wf_hh,
    const _Float16* __restrict__ Wr_ih, const _Float16* __restrict__ Wr_hh,
    float* __restrict__ pre_out)
{
  __shared__ _Float16 hb[8][16][32];   // per-wave h state (f16)

  const int lane = threadIdx.x & 31;
  const int wave = threadIdx.x >> 5;
  const int dir  = blockIdx.y;
  const int base = (blockIdx.x * 8 + wave) * 16;
  const int th   = lane >> 4;
  const int nlo  = lane & 15;

  const _Float16* Wih = dir ? Wr_ih : Wf_ih;
  const _Float16* Whh = dir ? Wr_hh : Wf_hh;
  const float* bih = dir ? b_ih_r : b_ih_f;
  const float* bhh = dir ? b_hh_r : b_hh_f;

  float bias[8];
#pragma unroll
  for (int nt = 0; nt < 8; ++nt)
    bias[nt] = bih[nlo + nt * 16] + bhh[nlo + nt * 16];

  for (int i = lane; i < 16 * 32; i += 32)
    hb[wave][i >> 5][i & 31] = (_Float16)0.0f;

  float c[2][8];
#pragma unroll
  for (int jh = 0; jh < 2; ++jh)
#pragma unroll
    for (int r = 0; r < 8; ++r) c[jh][r] = 0.0f;

  // this lane's x row base (row m = lane&15 of the batch tile)
  const _Float16* xbase = X16 + (size_t)(base + nlo) * TXn * 64;

  for (int t = 0; t < TXn; ++t) {
    const int tt = dir ? (TXn - 1 - t) : t;
    const _Float16* xrow = xbase + (size_t)tt * 64;

    h16 a0 = frag_a_glb(xrow, 0,  lane);
    h16 a1 = frag_a_glb(xrow, 32, lane);
    h16 ah = frag_a_lds(&hb[wave][0][0], 32, 0, lane);

#pragma unroll
    for (int jh = 0; jh < 2; ++jh) {
      // gate tiles for this j-group: i,f,g,o at n-tiles jh, 2+jh, 4+jh, 6+jh
      f8 acc[4];
#pragma unroll
      for (int q = 0; q < 4; ++q) {
        float bv = bias[2 * q + jh];
#pragma unroll
        for (int r = 0; r < 8; ++r) acc[q][r] = bv;
      }
#pragma unroll
      for (int q = 0; q < 4; ++q) {
        int n = nlo + (2 * q + jh) * 16;
        acc[q] = wmma16(a0, frag_b_glb(Wih + (size_t)n * 64, 0, lane), acc[q]);
      }
#pragma unroll
      for (int q = 0; q < 4; ++q) {
        int n = nlo + (2 * q + jh) * 16;
        acc[q] = wmma16(a1, frag_b_glb(Wih + (size_t)n * 64, 32, lane), acc[q]);
      }
#pragma unroll
      for (int q = 0; q < 4; ++q) {
        int n = nlo + (2 * q + jh) * 16;
        acc[q] = wmma16(ah, frag_b_glb(Whh + (size_t)n * 32, 0, lane), acc[q]);
      }
      // LSTM cell: lane owns unit j = nlo + 16*jh for rows m = r + 8*th
      int j = nlo + 16 * jh;
#pragma unroll
      for (int r = 0; r < 8; ++r) {
        int m = r + 8 * th;
        float ig = sigf(acc[0][r]);
        float fg = sigf(acc[1][r]);
        float gg = tanh_fast(acc[2][r]);
        float og = sigf(acc[3][r]);
        float cn = fg * c[jh][r] + ig * gg;
        c[jh][r] = cn;
        float h = og * tanh_fast(cn);
        hb[wave][m][j] = (_Float16)h;
        pre_out[((size_t)(base + m) * TXn + tt) * Hn + dir * Dn + j] = h;
      }
    }
  }
}

// ---------------------------------------------------------------------------
// K2: E1 = pre_out @ w_a1[:, :64]^T   (decoder-step invariant attention term)
// ---------------------------------------------------------------------------
__global__ void __launch_bounds__(256) attn_pre_kernel(
    const float* __restrict__ pre_out, const _Float16* __restrict__ Wa1x,
    float* __restrict__ e1)
{
  __shared__ _Float16 ps[8][16][64];
  const int lane = threadIdx.x & 31;
  const int wave = threadIdx.x >> 5;
  const int row0 = (blockIdx.x * 8 + wave) * 16;
  const int th   = lane >> 4;
  const int n    = lane & 15;

  for (int i = lane; i < 16 * 64; i += 32) {
    int m = i >> 6, k = i & 63;
    ps[wave][m][k] = (_Float16)pre_out[(size_t)(row0 + m) * Hn + k];
  }
  h16 a0 = frag_a_lds(&ps[wave][0][0], 64, 0,  lane);
  h16 a1 = frag_a_lds(&ps[wave][0][0], 64, 32, lane);
  f8 acc;
#pragma unroll
  for (int r = 0; r < 8; ++r) acc[r] = 0.0f;
  acc = wmma16(a0, frag_b_glb(Wa1x + n * 64, 0,  lane), acc);
  acc = wmma16(a1, frag_b_glb(Wa1x + n * 64, 32, lane), acc);
#pragma unroll
  for (int r = 0; r < 8; ++r)
    e1[(size_t)(row0 + r + 8 * th) * 16 + n] = acc[r];
}

// ---------------------------------------------------------------------------
// K3: attention decoder, 10 sequential steps, 16 batch rows per wave.
// Gate tiles processed per j-group (jh, 4+jh, 8+jh, 12+jh) -> 4 live acc frags.
// ---------------------------------------------------------------------------
__global__ void __launch_bounds__(256) decoder_kernel(
    const float* __restrict__ pre_out, const float* __restrict__ e1,
    const _Float16* __restrict__ Wa1s, const _Float16* __restrict__ Wp,
    const float* __restrict__ b_a1, const float* __restrict__ w_a2,
    const float* __restrict__ b_a2,
    const float* __restrict__ b_ih_p, const float* __restrict__ b_hh_p,
    const float* __restrict__ w_o, const float* __restrict__ b_o,
    float* __restrict__ logits)
{
  __shared__ _Float16 sbuf[8][16][64];   // decoder hidden s (f16)
  __shared__ _Float16 ctx [8][16][64];   // attention context (f16)
  __shared__ float    ebuf[8][16][32];   // attention energies / alphas
  __shared__ float    s1l [8][16][16];   // s @ w_a1[:,64:]^T

  const int lane = threadIdx.x & 31;
  const int wave = threadIdx.x >> 5;
  const int base = (blockIdx.x * 8 + wave) * 16;
  const int th   = lane >> 4;
  const int nlo  = lane & 15;

  float ba1r[An], wa2r[An];
  for (int a = 0; a < An; ++a) { ba1r[a] = b_a1[a]; wa2r[a] = w_a2[a]; }
  const float ba2 = b_a2[0];

  float bias[16];
#pragma unroll
  for (int nt = 0; nt < 16; ++nt)
    bias[nt] = b_ih_p[nlo + nt * 16] + b_hh_p[nlo + nt * 16];

  for (int i = lane; i < 16 * 64; i += 32)
    sbuf[wave][i >> 6][i & 63] = (_Float16)0.0f;

  float c[4][8];
#pragma unroll
  for (int jh = 0; jh < 4; ++jh)
#pragma unroll
    for (int r = 0; r < 8; ++r) c[jh][r] = 0.0f;

  for (int ty = 0; ty < TYn; ++ty) {
    // ---- S1 = s @ w_a1[:,64:]^T (one 16-col tile, padded 10->16) ----
    h16 as0 = frag_a_lds(&sbuf[wave][0][0], 64, 0,  lane);
    h16 as1 = frag_a_lds(&sbuf[wave][0][0], 64, 32, lane);
    {
      f8 s1a;
#pragma unroll
      for (int r = 0; r < 8; ++r) s1a[r] = 0.0f;
      s1a = wmma16(as0, frag_b_glb(Wa1s + nlo * 64, 0,  lane), s1a);
      s1a = wmma16(as1, frag_b_glb(Wa1s + nlo * 64, 32, lane), s1a);
#pragma unroll
      for (int r = 0; r < 8; ++r)
        s1l[wave][r + 8 * th][nlo] = s1a[r];
    }

    // ---- attention energies e[b][t] = relu(tanh(.)·w_a2 + b_a2) ----
    {
      int m = nlo;
      for (int t = th; t < TXn; t += 2) {
        const float* ep = e1 + ((size_t)(base + m) * TXn + t) * 16;
        float s = ba2;
        for (int a = 0; a < An; ++a)
          s += wa2r[a] * tanh_fast(ep[a] + s1l[wave][m][a] + ba1r[a]);
        ebuf[wave][m][t] = fmaxf(s, 0.0f);
      }
    }
    // ---- softmax over Tx (lower half-lanes, one batch row each) ----
    if (th == 0) {
      int m = nlo;
      float mx = -1e30f;
      for (int t = 0; t < TXn; ++t) mx = fmaxf(mx, ebuf[wave][m][t]);
      float sum = 0.0f;
      for (int t = 0; t < TXn; ++t) {
        float e = __expf(ebuf[wave][m][t] - mx);
        ebuf[wave][m][t] = e;
        sum += e;
      }
      float inv = __builtin_amdgcn_rcpf(sum);
      for (int t = 0; t < TXn; ++t) ebuf[wave][m][t] *= inv;
    }
    // ---- context = sum_t alpha * pre_out (float4 L2-streaming) ----
    {
      int m = nlo;
      const float* prow = pre_out + (size_t)(base + m) * TXn * Hn + th * 32;
      float4 s4[8];
#pragma unroll
      for (int q = 0; q < 8; ++q) s4[q] = make_float4(0.f, 0.f, 0.f, 0.f);
      for (int t = 0; t < TXn; ++t) {
        float a = ebuf[wave][m][t];
        const float4* p4 = (const float4*)(prow + (size_t)t * Hn);
#pragma unroll
        for (int q = 0; q < 8; ++q) {
          float4 v = p4[q];
          s4[q].x += a * v.x; s4[q].y += a * v.y;
          s4[q].z += a * v.z; s4[q].w += a * v.w;
        }
      }
#pragma unroll
      for (int q = 0; q < 8; ++q) {
        int hcol = th * 32 + q * 4;
        ctx[wave][m][hcol + 0] = (_Float16)s4[q].x;
        ctx[wave][m][hcol + 1] = (_Float16)s4[q].y;
        ctx[wave][m][hcol + 2] = (_Float16)s4[q].z;
        ctx[wave][m][hcol + 3] = (_Float16)s4[q].w;
      }
    }
    // ---- gates + cell update, one j-group at a time ----
    h16 ac0 = frag_a_lds(&ctx[wave][0][0], 64, 0,  lane);
    h16 ac1 = frag_a_lds(&ctx[wave][0][0], 64, 32, lane);
#pragma unroll
    for (int jh = 0; jh < 4; ++jh) {
      // i,f,g,o at n-tiles jh, 4+jh, 8+jh, 12+jh (same lanes in C layout)
      f8 acc[4];
#pragma unroll
      for (int q = 0; q < 4; ++q) {
        float bv = bias[4 * q + jh];
#pragma unroll
        for (int r = 0; r < 8; ++r) acc[q][r] = bv;
      }
#pragma unroll
      for (int q = 0; q < 4; ++q) {
        const _Float16* wrow = Wp + (size_t)(nlo + (4 * q + jh) * 16) * 128;
        acc[q] = wmma16(ac0, frag_b_glb(wrow, 0, lane), acc[q]);
      }
#pragma unroll
      for (int q = 0; q < 4; ++q) {
        const _Float16* wrow = Wp + (size_t)(nlo + (4 * q + jh) * 16) * 128;
        acc[q] = wmma16(ac1, frag_b_glb(wrow, 32, lane), acc[q]);
      }
#pragma unroll
      for (int q = 0; q < 4; ++q) {
        const _Float16* wrow = Wp + (size_t)(nlo + (4 * q + jh) * 16) * 128;
        acc[q] = wmma16(as0, frag_b_glb(wrow, 64, lane), acc[q]);
      }
#pragma unroll
      for (int q = 0; q < 4; ++q) {
        const _Float16* wrow = Wp + (size_t)(nlo + (4 * q + jh) * 16) * 128;
        acc[q] = wmma16(as1, frag_b_glb(wrow, 96, lane), acc[q]);
      }
      int j = nlo + 16 * jh;
#pragma unroll
      for (int r = 0; r < 8; ++r) {
        float ig = sigf(acc[0][r]);
        float fg = sigf(acc[1][r]);
        float gg = tanh_fast(acc[2][r]);
        float og = sigf(acc[3][r]);
        float cn = fg * c[jh][r] + ig * gg;
        c[jh][r] = cn;
        float s = og * tanh_fast(cn);
        sbuf[wave][r + 8 * th][j] = (_Float16)s;
      }
    }
    // ---- logits = s_new @ w_o^T + b_o ----
    {
      int m = nlo;
      for (int v = th; v < VOUT; v += 2) {
        float s = b_o[v];
        for (int j = 0; j < Hn; ++j)
          s += (float)sbuf[wave][m][j] * w_o[v * Hn + j];
        logits[((size_t)(base + m) * TYn + ty) * VOUT + v] = s;
      }
    }
  }
}

// ---------------------------------------------------------------------------
// K4: softmax over BATCH axis per (ty, v) column (reference quirk).
// ---------------------------------------------------------------------------
__global__ void __launch_bounds__(256) batch_softmax_kernel(
    const float* __restrict__ logits, float* __restrict__ out)
{
  __shared__ float red[256];
  const int tid = threadIdx.x;
  const int col = blockIdx.x;            // ty*VOUT + v
  const int stride = TYn * VOUT;         // 110

  float mx = -1e30f;
  for (int b = tid; b < Bn; b += 256)
    mx = fmaxf(mx, logits[(size_t)b * stride + col]);
  red[tid] = mx;
  __syncthreads();
  for (int s = 128; s > 0; s >>= 1) {
    if (tid < s) red[tid] = fmaxf(red[tid], red[tid + s]);
    __syncthreads();
  }
  mx = red[0];
  __syncthreads();

  float sum = 0.0f;
  for (int b = tid; b < Bn; b += 256)
    sum += __expf(logits[(size_t)b * stride + col] - mx);
  red[tid] = sum;
  __syncthreads();
  for (int s = 128; s > 0; s >>= 1) {
    if (tid < s) red[tid] += red[tid + s];
    __syncthreads();
  }
  float inv = 1.0f / red[0];

  for (int b = tid; b < Bn; b += 256)
    out[(size_t)b * stride + col] =
        __expf(logits[(size_t)b * stride + col] - mx) * inv;
}

// ---------------------------------------------------------------------------
extern "C" void kernel_launch(void* const* d_in, const int* in_sizes, int n_in,
                              void* d_out, int out_size, void* d_ws, size_t ws_size,
                              hipStream_t stream) {
  (void)in_sizes; (void)n_in; (void)out_size; (void)ws_size;

  const float* X      = (const float*)d_in[0];
  const float* w_ih_f = (const float*)d_in[1];
  const float* w_hh_f = (const float*)d_in[2];
  const float* b_ih_f = (const float*)d_in[3];
  const float* b_hh_f = (const float*)d_in[4];
  const float* w_ih_r = (const float*)d_in[5];
  const float* w_hh_r = (const float*)d_in[6];
  const float* b_ih_r = (const float*)d_in[7];
  const float* b_hh_r = (const float*)d_in[8];
  const float* w_ih_p = (const float*)d_in[9];
  const float* w_hh_p = (const float*)d_in[10];
  const float* b_ih_p = (const float*)d_in[11];
  const float* b_hh_p = (const float*)d_in[12];
  const float* w_a1   = (const float*)d_in[13];
  const float* b_a1   = (const float*)d_in[14];
  const float* w_a2   = (const float*)d_in[15];
  const float* b_a2   = (const float*)d_in[16];
  const float* w_o    = (const float*)d_in[17];
  const float* b_o    = (const float*)d_in[18];

  char* ws = (char*)d_ws;
  const size_t PRE_B = (size_t)Bn * TXn * Hn * 4;     // 125,829,120
  const size_t E1_B  = (size_t)Bn * TXn * 16 * 4;     //  31,457,280
  const size_t LGT_B = (size_t)Bn * TYn * VOUT * 4;   //   7,208,960
  const size_t X16_B = (size_t)Bn * TXn * 64 * 2;     //  62,914,560

  float*    pre = (float*)(ws);
  float*    e1  = (float*)(ws + PRE_B);
  float*    lgt = (float*)(ws + PRE_B + E1_B);
  _Float16* x16 = (_Float16*)(ws + PRE_B + E1_B + LGT_B);
  char*     wb  = ws + PRE_B + E1_B + LGT_B + X16_B;

  _Float16* Wf_ih = (_Float16*)(wb);                  // 128*64
  _Float16* Wf_hh = (_Float16*)(wb + 16384);          // 128*32
  _Float16* Wr_ih = (_Float16*)(wb + 24576);
  _Float16* Wr_hh = (_Float16*)(wb + 40960);
  _Float16* Wp    = (_Float16*)(wb + 49152);          // 256*128
  _Float16* Wa1x  = (_Float16*)(wb + 114688);         // 16*64
  _Float16* Wa1s  = (_Float16*)(wb + 116736);         // 16*64

  pack_weights_kernel<<<1, 256, 0, stream>>>(
      w_ih_f, w_hh_f, w_ih_r, w_hh_r, w_ih_p, w_hh_p, w_a1,
      Wf_ih, Wf_hh, Wr_ih, Wr_hh, Wp, Wa1x, Wa1s);

  convert_x_kernel<<<(Bn * TXn * 64) / 256, 256, 0, stream>>>(X, x16);

  prelstm_kernel<<<dim3(Bn / 128, 2), 256, 0, stream>>>(
      x16, b_ih_f, b_hh_f, b_ih_r, b_hh_r, Wf_ih, Wf_hh, Wr_ih, Wr_hh, pre);

  attn_pre_kernel<<<(Bn * TXn) / 128, 256, 0, stream>>>(pre, Wa1x, e1);

  decoder_kernel<<<Bn / 128, 256, 0, stream>>>(
      pre, e1, Wa1s, Wp, b_a1, w_a2, b_a2, b_ih_p, b_hh_p, w_o, b_o, lgt);

  batch_softmax_kernel<<<TYn * VOUT, 256, 0, stream>>>(lgt, (float*)d_out);
}
